// double_conv2d_bn_mamba_19000935317605
// MI455X (gfx1250) — matrix operand order repaired and checked by reading, not compiled, not run on hardware
//
#include <hip/hip_runtime.h>
#include <math.h>
#include <stdint.h>

// Problem constants (from reference)
#define HH   64
#define WW_  64
#define HWs  4096
#define BB   4
#define CC   64
#define LL   4096
#define DIi  128
#define DSs  16
#define DRr  4
#define KK   4

typedef __attribute__((ext_vector_type(16))) _Float16 v16h;
typedef __attribute__((ext_vector_type(8)))  float    v8f;

union F16x16 { v16h v; _Float16 h[16]; };
union F32x8  { v8f  v; float    f[8];  };

__device__ __forceinline__ float siluf(float x)     { return x / (1.0f + __expf(-x)); }
__device__ __forceinline__ float softplusf(float x) { return (x > 20.0f) ? x : log1pf(__expf(x)); }

// Direction index remap: scan position t (for direction k) -> hw-flat spatial index.
// k=0: row-major; k=1: col-major; k=2: reversed row-major; k=3: reversed col-major.
__device__ __forceinline__ int dirmap(int k, int t) {
  switch (k & 3) {
    case 0: return t;
    case 1: return ((t & 63) << 6) | (t >> 6);
    case 2: return (LL - 1) - t;
    default: { int tp = (LL - 1) - t; return ((tp & 63) << 6) | (tp >> 6); }
  }
}

// ---------------------------------------------------------------------------
// Implicit-GEMM 3x3 conv, NCHW, 64->64 channels, SAME padding, WMMA f16->f32.
// One block per (b, h) image row: M tile = 64 pixels (w), N = 64 couts,
// K = 9 taps * 64 cin, consumed in 32-wide chunks.
// Boundary handling is branch-free: clamped address + cndmask select.
// ---------------------------------------------------------------------------
__global__ __launch_bounds__(256) void conv3x3_wmma_k(
    const float* __restrict__ X, const float* __restrict__ Wt,
    const float* __restrict__ bias, float* __restrict__ Y,
    const float* __restrict__ residual)
{
  __shared__ _Float16 As[64][32];
  __shared__ _Float16 Bs[64][32];

  const int blk  = blockIdx.x;       // 0..B*H-1
  const int b    = blk >> 6;
  const int h    = blk & 63;
  const int tid  = threadIdx.x;
  const int lane = tid & 31;
  const int wid  = tid >> 5;
  const int wm   = wid & 3;          // M sub-tile (16 rows)
  const int wn   = wid >> 2;         // N sub-tile (32 cols)

  v8f acc0 = {0.f,0.f,0.f,0.f,0.f,0.f,0.f,0.f};
  v8f acc1 = {0.f,0.f,0.f,0.f,0.f,0.f,0.f,0.f};

  for (int tap = 0; tap < 9; ++tap) {
    const int ky  = tap / 3 - 1;
    const int kx  = tap % 3 - 1;
    const int hh2 = h + ky;
    const bool hok = (hh2 >= 0) && (hh2 < HH);
    const int hc  = hh2 < 0 ? 0 : (hh2 > HH - 1 ? HH - 1 : hh2);
    for (int cc = 0; cc < 2; ++cc) {
      const int c0 = cc * 32;
      // A tile: 64 pixels x 32 cin (row-fast => coalesced over w), branch-free pad
      #pragma unroll
      for (int i = 0; i < 8; ++i) {
        int e   = tid + i * 256;
        int kk  = e >> 6;
        int row = e & 63;
        int ww2 = row + kx;
        bool ok = hok && (ww2 >= 0) && (ww2 < WW_);
        int wc  = ww2 < 0 ? 0 : (ww2 > WW_ - 1 ? WW_ - 1 : ww2);
        float v = X[(size_t)(b * CC + c0 + kk) * HWs + hc * WW_ + wc];
        As[row][kk] = (_Float16)(ok ? v : 0.0f);
      }
      // B tile: 64 couts x 32 cin for this tap
      #pragma unroll
      for (int i = 0; i < 8; ++i) {
        int e  = tid + i * 256;
        int kk = e & 31;
        int n  = e >> 5;
        Bs[n][kk] = (_Float16)Wt[(size_t)(n * CC + c0 + kk) * 9 + tap];
      }
      __syncthreads();

      // Pack fragments in contiguous k order (k = 16*half + e), same for A and B (wave32).
      F16x16 a, b0, b1;
      const int r     = lane & 15;
      const int abase = (lane < 16) ? 0 : 16;
      #pragma unroll
      for (int e = 0; e < 16; ++e) {
        int kidx = abase + e;
        a.h[e] = As[16 * wm + r][kidx];
      }
      const int kb = (lane < 16) ? 0 : 16;
      #pragma unroll
      for (int e = 0; e < 16; ++e) {
        b0.h[e] = Bs[32 * wn + r][kb + e];
        b1.h[e] = Bs[32 * wn + 16 + r][kb + e];
      }
      acc0 = __builtin_amdgcn_wmma_f32_16x16x32_f16(false, a.v, false, b0.v, (short)0, acc0, false, false);
      acc1 = __builtin_amdgcn_wmma_f32_16x16x32_f16(false, a.v, false, b1.v, (short)0, acc1, false, false);
      __syncthreads();
    }
  }

  // Epilogue: C/D layout -> (w pixel, cout). Add bias (+ optional residual).
  F32x8 r0, r1; r0.v = acc0; r1.v = acc1;
  const int nlo   = lane & 15;
  const int mhalf = (lane < 16) ? 0 : 8;
  #pragma unroll
  for (int rr = 0; rr < 8; ++rr) {
    int w  = 16 * wm + mhalf + rr;
    int n0 = 32 * wn + nlo;
    int n1 = n0 + 16;
    size_t o0 = (size_t)(b * CC + n0) * HWs + h * WW_ + w;
    size_t o1 = (size_t)(b * CC + n1) * HWs + h * WW_ + w;
    float v0 = r0.f[rr] + bias[n0];
    float v1 = r1.f[rr] + bias[n1];
    if (residual) { v0 += residual[o0]; v1 += residual[o1]; }
    Y[o0] = v0;
    Y[o1] = v1;
  }
}

// ---------------------------------------------------------------------------
// Generic NT GEMM: C[M,N] = A[M,K] * W[N,K]^T, f32 in/out, f16 WMMA multiply.
// 64x64 tile per block, K consumed in 32-chunks. M,N multiples of 64; K of 32.
// Tiles are staged to LDS with the gfx1250 async global->LDS path
// (global_load_async_to_lds_b32, ASYNCcnt), converted f32->f16 at pack time.
// ---------------------------------------------------------------------------
__global__ __launch_bounds__(256) void gemm_nt_wmma_k(
    const float* __restrict__ A, const float* __restrict__ Wt,
    float* __restrict__ C, int M, int N, int K)
{
  __shared__ float Asf[64][32];
  __shared__ float Bsf[64][32];

  const int m0   = blockIdx.x * 64;
  const int n0   = blockIdx.y * 64;
  const int tid  = threadIdx.x;
  const int lane = tid & 31;
  const int wid  = tid >> 5;
  const int wm   = wid & 3;
  const int wn   = wid >> 2;

  v8f acc0 = {0.f,0.f,0.f,0.f,0.f,0.f,0.f,0.f};
  v8f acc1 = {0.f,0.f,0.f,0.f,0.f,0.f,0.f,0.f};

  for (int k0 = 0; k0 < K; k0 += 32) {
    // Async-stage A and B tiles into LDS (one 4B element per lane per issue;
    // LDS destination byte-offset rides in the VDST VGPR, wave-relative).
    #pragma unroll
    for (int i = 0; i < 8; ++i) {
      int e   = tid + i * 256;
      int kk  = e & 31;
      int row = e >> 5;
      uint32_t la = (uint32_t)(size_t)&Asf[row][kk];
      uint32_t lb = (uint32_t)(size_t)&Bsf[row][kk];
      uint64_t ga = (uint64_t)(size_t)&A [(size_t)(m0 + row) * K + k0 + kk];
      uint64_t gb = (uint64_t)(size_t)&Wt[(size_t)(n0 + row) * K + k0 + kk];
      asm volatile("global_load_async_to_lds_b32 %0, %1, off"
                   :: "v"(la), "v"(ga) : "memory");
      asm volatile("global_load_async_to_lds_b32 %0, %1, off"
                   :: "v"(lb), "v"(gb) : "memory");
    }
    asm volatile("s_wait_asynccnt 0x0" ::: "memory");
    __syncthreads();

    F16x16 a, b0, b1;
    const int r     = lane & 15;
    const int abase = (lane < 16) ? 0 : 16;
    #pragma unroll
    for (int e = 0; e < 16; ++e) {
      int kidx = abase + e;
      a.h[e] = (_Float16)Asf[16 * wm + r][kidx];
    }
    const int kb = (lane < 16) ? 0 : 16;
    #pragma unroll
    for (int e = 0; e < 16; ++e) {
      b0.h[e] = (_Float16)Bsf[32 * wn + r][kb + e];
      b1.h[e] = (_Float16)Bsf[32 * wn + 16 + r][kb + e];
    }
    acc0 = __builtin_amdgcn_wmma_f32_16x16x32_f16(false, a.v, false, b0.v, (short)0, acc0, false, false);
    acc1 = __builtin_amdgcn_wmma_f32_16x16x32_f16(false, a.v, false, b1.v, (short)0, acc1, false, false);
    __syncthreads();
  }

  F32x8 r0, r1; r0.v = acc0; r1.v = acc1;
  const int nlo   = lane & 15;
  const int mhalf = (lane < 16) ? 0 : 8;
  #pragma unroll
  for (int rr = 0; rr < 8; ++rr) {
    int m = m0 + 16 * wm + mhalf + rr;
    int n = n0 + 32 * wn + nlo;
    C[(size_t)m * N + n]      = r0.f[rr];
    C[(size_t)m * N + n + 16] = r1.f[rr];
  }
}

// ---------------------------------------------------------------------------
// Instance norm over (H,W) per (b,c) + LeakyReLU(0.2). One block per (b,c).
// ---------------------------------------------------------------------------
__global__ __launch_bounds__(256) void inorm_lrelu_k(
    const float* __restrict__ X, float* __restrict__ Y)
{
  __shared__ float ssum[256];
  __shared__ float ssq[256];
  const int bc = blockIdx.x;
  const float* xr = X + (size_t)bc * HWs;
  float s = 0.f, q = 0.f;
  for (int i = threadIdx.x; i < HWs; i += 256) { float v = xr[i]; s += v; q += v * v; }
  ssum[threadIdx.x] = s; ssq[threadIdx.x] = q;
  __syncthreads();
  for (int off = 128; off > 0; off >>= 1) {
    if ((int)threadIdx.x < off) {
      ssum[threadIdx.x] += ssum[threadIdx.x + off];
      ssq [threadIdx.x] += ssq [threadIdx.x + off];
    }
    __syncthreads();
  }
  const float mean = ssum[0] * (1.0f / HWs);
  const float var  = ssq[0] * (1.0f / HWs) - mean * mean;
  const float rs   = rsqrtf(var + 1e-5f);
  float* yr = Y + (size_t)bc * HWs;
  for (int i = threadIdx.x; i < HWs; i += 256) {
    float v = (xr[i] - mean) * rs;
    yr[i] = (v >= 0.f) ? v : 0.2f * v;
  }
}

// LayerNorm over channels of NCHW x1 -> NHWC row (B*L, 64), with gamma/beta.
__global__ void ln_ch_k(const float* __restrict__ Xn, const float* __restrict__ g,
                        const float* __restrict__ be, float* __restrict__ Yn)
{
  int row = blockIdx.x * blockDim.x + threadIdx.x;   // b*L + l
  if (row >= BB * LL) return;
  int b = row >> 12, l = row & (LL - 1);
  const float* base = Xn + (size_t)b * CC * HWs + l;
  float s = 0.f, q = 0.f;
  for (int c = 0; c < CC; ++c) { float v = base[(size_t)c * HWs]; s += v; q += v * v; }
  float mean = s * (1.0f / CC);
  float var  = q * (1.0f / CC) - mean * mean;
  float rs   = rsqrtf(var + 1e-5f);
  float* yr = Yn + (size_t)row * CC;
  for (int c = 0; c < CC; ++c)
    yr[c] = (base[(size_t)c * HWs] - mean) * rs * g[c] + be[c];
}

// Depthwise 3x3 conv on xz[:, :, 0:128] (layout (B,L,256)) + SiLU -> xin (B,L,128)
__global__ void dwconv_silu_k(const float* __restrict__ xz, const float* __restrict__ w9,
                              const float* __restrict__ bso, float* __restrict__ xin)
{
  int idx = blockIdx.x * blockDim.x + threadIdx.x;   // ((b*L)+l)*DI + d
  if (idx >= BB * LL * DIi) return;
  int d = idx & (DIi - 1);
  int l = (idx >> 7) & (LL - 1);
  int b = idx >> 19;
  int h = l >> 6, w = l & 63;
  float acc = bso[d];
  #pragma unroll
  for (int tap = 0; tap < 9; ++tap) {
    int hh2 = h + tap / 3 - 1;
    int ww2 = w + tap % 3 - 1;
    if (hh2 >= 0 && hh2 < HH && ww2 >= 0 && ww2 < WW_)
      acc += w9[d * 9 + tap] * xz[(size_t)(b * LL + hh2 * WW_ + ww2) * 256 + d];
  }
  xin[(size_t)idx] = siluf(acc);
}

// x_dbl[b,k,l,c] = sum_d x_proj_w[k,c,d] * xin[b, dirmap(k,l), d]   (c = 0..35)
__global__ void xproj_k(const float* __restrict__ xin, const float* __restrict__ xpw,
                        float* __restrict__ xdbl)
{
  int idx = blockIdx.x * blockDim.x + threadIdx.x;
  if (idx >= BB * KK * LL * 36) return;
  int c   = idx % 36;
  int rem = idx / 36;                  // b*K*L + k*L + l
  int l   = rem & (LL - 1);
  int k   = (rem >> 12) & 3;
  int b   = rem >> 14;
  int m   = dirmap(k, l);
  const float* xr = xin + (size_t)(b * LL + m) * DIi;
  const float* wr = xpw + (size_t)(k * 36 + c) * DIi;
  float acc = 0.f;
  #pragma unroll 4
  for (int d = 0; d < DIi; ++d) acc += xr[d] * wr[d];
  xdbl[idx] = acc;
}

__global__ void zero_k(float* __restrict__ p, int n)
{
  int i = blockIdx.x * blockDim.x + threadIdx.x;
  if (i < n) p[i] = 0.f;
}

// Selective scan: one block per (b,k); thread = channel d; 16-state SSM in regs.
// Per step: dt = softplus(dt_proj . dts + b); h = h*exp(dt*A) + dt*u*B; y = h.C.
// Accumulates y + Ds*u into y_comb[b, dirmap(k,t), d] (4 directions via atomics).
__global__ __launch_bounds__(128) void scan_k(
    const float* __restrict__ xdbl, const float* __restrict__ xin,
    const float* __restrict__ dtw, const float* __restrict__ dtb,
    const float* __restrict__ alogs, const float* __restrict__ dsv,
    float* __restrict__ ycomb)
{
  __shared__ float sx[36];
  const int bk = blockIdx.x;
  const int b  = bk >> 2;
  const int k  = bk & 3;
  const int d  = threadIdx.x;

  float wdt[DRr];
  #pragma unroll
  for (int r2 = 0; r2 < DRr; ++r2) wdt[r2] = dtw[(size_t)(k * DIi + d) * DRr + r2];
  const float db = dtb[k * DIi + d];
  float Aneg[DSs];
  #pragma unroll
  for (int n = 0; n < DSs; ++n) Aneg[n] = -__expf(alogs[(size_t)(k * DIi + d) * DSs + n]);
  const float Dv = dsv[k * DIi + d];
  float hs[DSs];
  #pragma unroll
  for (int n = 0; n < DSs; ++n) hs[n] = 0.f;

  const float* xd = xdbl + (size_t)(b * KK + k) * LL * 36;
  for (int t = 0; t < LL; ++t) {
    if (d < 36) sx[d] = xd[(size_t)t * 36 + d];
    __syncthreads();
    float dtv = db;
    #pragma unroll
    for (int r2 = 0; r2 < DRr; ++r2) dtv += wdt[r2] * sx[r2];
    dtv = softplusf(dtv);
    const int   m  = dirmap(k, t);
    const float u  = xin[(size_t)(b * LL + m) * DIi + d];
    const float du = dtv * u;
    float y = 0.f;
    #pragma unroll
    for (int n = 0; n < DSs; ++n) {
      float dec = __expf(dtv * Aneg[n]);
      hs[n] = hs[n] * dec + du * sx[4 + n];
      y += hs[n] * sx[20 + n];
    }
    y += Dv * u;
    atomicAdd(&ycomb[(size_t)(b * LL + m) * DIi + d], y);
    __syncthreads();
  }
}

// out-norm LN over DI, then gate with silu(z); in-place on y_comb (B,L,DI).
__global__ void lngate_k(float* __restrict__ yc, const float* __restrict__ xz,
                         const float* __restrict__ g, const float* __restrict__ be)
{
  int row = blockIdx.x * blockDim.x + threadIdx.x;
  if (row >= BB * LL) return;
  float* yr = yc + (size_t)row * DIi;
  const float* zr = xz + (size_t)row * 256 + DIi;
  float s = 0.f, q = 0.f;
  for (int d = 0; d < DIi; ++d) { float v = yr[d]; s += v; q += v * v; }
  float mean = s * (1.0f / DIi);
  float var  = q * (1.0f / DIi) - mean * mean;
  float rs   = rsqrtf(var + 1e-5f);
  for (int d = 0; d < DIi; ++d) {
    float v = (yr[d] - mean) * rs * g[d] + be[d];
    yr[d] = v * siluf(zr[d]);
  }
}

// res (NCHW) = x1 * skip_scale + out_ss (NHWC -> transpose read)
__global__ void resmix_k(const float* __restrict__ x1, const float* __restrict__ oss,
                         const float* __restrict__ skip, float* __restrict__ res)
{
  int idx = blockIdx.x * blockDim.x + threadIdx.x;   // (b*64+c)*HW + l
  if (idx >= BB * CC * HWs) return;
  int l = idx & (HWs - 1);
  int c = (idx >> 12) & 63;
  int b = idx >> 18;
  res[idx] = x1[idx] * skip[0] + oss[(size_t)(b * HWs + l) * CC + c];
}

// ---------------------------------------------------------------------------
extern "C" void kernel_launch(void* const* d_in, const int* in_sizes, int n_in,
                              void* d_out, int out_size, void* d_ws, size_t ws_size,
                              hipStream_t stream)
{
  (void)in_sizes; (void)n_in; (void)out_size; (void)ws_size;
  const float* x          = (const float*)d_in[0];
  const float* conv1_w    = (const float*)d_in[1];
  const float* conv1_b    = (const float*)d_in[2];
  const float* conv2_w    = (const float*)d_in[3];
  const float* conv2_b    = (const float*)d_in[4];
  const float* cf_w       = (const float*)d_in[5];
  const float* cf_b       = (const float*)d_in[6];
  const float* ln_g       = (const float*)d_in[7];
  const float* ln_b       = (const float*)d_in[8];
  const float* in_proj_w  = (const float*)d_in[9];
  const float* dw_w       = (const float*)d_in[10];
  const float* dw_b       = (const float*)d_in[11];
  const float* x_proj_w   = (const float*)d_in[12];
  const float* dt_proj_w  = (const float*)d_in[13];
  const float* dt_proj_b  = (const float*)d_in[14];
  const float* A_logs     = (const float*)d_in[15];
  const float* Ds         = (const float*)d_in[16];
  const float* out_norm_g = (const float*)d_in[17];
  const float* out_norm_b = (const float*)d_in[18];
  const float* out_proj_w = (const float*)d_in[19];
  const float* skip       = (const float*)d_in[20];
  const float* cb_w       = (const float*)d_in[21];
  const float* cb_b       = (const float*)d_in[22];
  float* out = (float*)d_out;
  float* ws  = (float*)d_ws;

  const size_t S_IMG = (size_t)BB * CC * HWs;        // 1,048,576 floats
  float* bufA = ws;                                  // conv1 out / normed
  float* bufB = bufA + S_IMG;                        // conv2 out, later cb out
  float* x1   = bufB + S_IMG;                        // cf conv out
  float* xsln = x1   + S_IMG;                        // LN(xs) NHWC (B*L,64)
  float* xz   = xsln + S_IMG;                        // in_proj out (B*L,256)
  float* xin  = xz   + 4 * S_IMG;                    // dwconv+silu (B*L,128)
  float* xdbl = xin  + 2 * S_IMG;                    // x_proj out (B,K,L,36)
  float* yc   = xdbl + (size_t)BB * KK * LL * 36;    // y_comb (B*L,128)
  float* oss  = yc   + (size_t)BB * LL * DIi;        // out_proj out (B*L,64)
  float* res  = oss  + S_IMG;                        // skip-mixed NCHW

  const dim3 blk256(256);
  const dim3 cgrid(BB * HH);                         // 256 blocks per conv

  // out = lrelu(inorm(conv1(x)))
  conv3x3_wmma_k<<<cgrid, blk256, 0, stream>>>(x, conv1_w, conv1_b, bufA, nullptr);
  inorm_lrelu_k<<<dim3(BB * CC), blk256, 0, stream>>>(bufA, bufA);
  // out = conv2(out)
  conv3x3_wmma_k<<<cgrid, blk256, 0, stream>>>(bufA, conv2_w, conv2_b, bufB, nullptr);
  // mamba: x1 = cf(out)
  conv3x3_wmma_k<<<cgrid, blk256, 0, stream>>>(bufB, cf_w, cf_b, x1, nullptr);
  // LN over channels -> NHWC
  ln_ch_k<<<dim3(64), blk256, 0, stream>>>(x1, ln_g, ln_b, xsln);
  // in_proj: (16384 x 64) * (256 x 64)^T
  gemm_nt_wmma_k<<<dim3(256, 4), blk256, 0, stream>>>(xsln, in_proj_w, xz, 16384, 256, 64);
  // depthwise conv + silu
  dwconv_silu_k<<<dim3(8192), blk256, 0, stream>>>(xz, dw_w, dw_b, xin);
  // x_proj for all 4 directions (flips/transposes via index math)
  xproj_k<<<dim3(9216), blk256, 0, stream>>>(xin, x_proj_w, xdbl);
  // selective scan (accumulates into yc)
  zero_k<<<dim3(8192), blk256, 0, stream>>>(yc, BB * LL * DIi);
  scan_k<<<dim3(BB * KK), dim3(128), 0, stream>>>(xdbl, xin, dt_proj_w, dt_proj_b,
                                                  A_logs, Ds, yc);
  // out-norm + silu(z) gate
  lngate_k<<<dim3(64), blk256, 0, stream>>>(yc, xz, out_norm_g, out_norm_b);
  // out_proj: (16384 x 128) * (64 x 128)^T
  gemm_nt_wmma_k<<<dim3(256, 1), blk256, 0, stream>>>(yc, out_proj_w, oss, 16384, 64, 128);
  // xs*skip + ss2d -> NCHW
  resmix_k<<<dim3(4096), blk256, 0, stream>>>(x1, oss, skip, res);
  // cb conv + x1 residual
  conv3x3_wmma_k<<<cgrid, blk256, 0, stream>>>(res, cb_w, cb_b, bufB, x1);
  // final inorm + lrelu -> d_out
  inorm_lrelu_k<<<dim3(BB * CC), blk256, 0, stream>>>(bufB, out);
}
